// CausalMultiHeadSelfAttention_22780506538607
// MI455X (gfx1250) — matrix operand order repaired and checked by reading, not compiled, not run on hardware
//
#include <hip/hip_runtime.h>
#include <hip/hip_bf16.h>
#include <stdint.h>

// Problem constants (CausalMultiHeadSelfAttention)
#define D_MODEL   1024
#define NUM_HEADS 16
#define DK        64
#define SEQ       2048
#define BATCH     4
#define M_TOTAL   (BATCH * SEQ)   // 8192 rows

typedef __attribute__((ext_vector_type(16))) __bf16       bf16x16;
typedef __attribute__((ext_vector_type(8)))  float        f32x8;
typedef __attribute__((ext_vector_type(4)))  unsigned int u32x4;
typedef __attribute__((ext_vector_type(4)))  int          i32x4;

union BF16Frag { bf16x16 v; u32x4 q[2]; };

__device__ __forceinline__ unsigned short f2bf(float f) {
    unsigned int u = __float_as_uint(f);
    u += 0x7FFFu + ((u >> 16) & 1u);   // round-to-nearest-even
    return (unsigned short)(u >> 16);
}

__device__ __forceinline__ u32x4 ld16B(const unsigned short* p) {
    return *reinterpret_cast<const u32x4*>(p);
}

// 16-byte global -> LDS copy.  Uses the CDNA5 async-to-LDS path (ASYNCcnt)
// when the toolchain exposes it; otherwise VGPR round-trip (LOADcnt+DScnt).
#if defined(__has_builtin)
#if __has_builtin(__builtin_amdgcn_global_load_async_to_lds_b128)
#define HAVE_ASYNC_LDS 1
#endif
#endif

__device__ __forceinline__ void copy16(const unsigned short* g, unsigned short* l) {
#ifdef HAVE_ASYNC_LDS
    __builtin_amdgcn_global_load_async_to_lds_b128(
        (__attribute__((address_space(1))) i32x4*)g,
        (__attribute__((address_space(3))) i32x4*)l, 0, 0);
#else
    *reinterpret_cast<u32x4*>(l) = *reinterpret_cast<const u32x4*>(g);
#endif
}

__device__ __forceinline__ void async_fence() {
#ifdef HAVE_ASYNC_LDS
    asm volatile("s_wait_asynccnt 0x0" ::: "memory");
#endif
}

// ---------------------------------------------------------------------------
// fp32 -> bf16 conversion (8 elements / thread, b128 stores)
// ---------------------------------------------------------------------------
__global__ void cvt_f32_bf16(const float* __restrict__ in,
                             unsigned short* __restrict__ out, int n) {
    int i = (blockIdx.x * blockDim.x + threadIdx.x) * 8;
    if (i >= n) return;
    u32x4 r;
#pragma unroll
    for (int k = 0; k < 4; ++k) {
        float a = in[i + 2 * k], b = in[i + 2 * k + 1];
        r[k] = (unsigned)f2bf(a) | ((unsigned)f2bf(b) << 16);
    }
    *reinterpret_cast<u32x4*>(out + i) = r;
}

// ---------------------------------------------------------------------------
// Y = A[M,1024] @ W^T, W[1024,1024] row-major (B column n == W row n).
// Block tile 128(M) x 128(N), BK = 32, double-buffered LDS staging.
// 8 waves, each computes a 32x64 sub-tile => 8 WMMAs per wave per K-step.
// mode 0: bf16 out [B,H,S,dk] | mode 1: bf16 out [B,H,dk,S] | mode 2: f32 row-major
// ---------------------------------------------------------------------------
__global__ void __launch_bounds__(256)
gemm_xwt(const unsigned short* __restrict__ A,
         const unsigned short* __restrict__ W,
         unsigned short* __restrict__ outb,
         float* __restrict__ outf,
         int mode) {
    __shared__ unsigned short As[2][128 * 32];   // 8 KB per stage
    __shared__ unsigned short Bs[2][128 * 32];   // 8 KB per stage

    const int tid  = threadIdx.x;
    const int lane = tid & 31;
    const int wav  = __builtin_amdgcn_readfirstlane(tid >> 5);
    const int l16  = lane & 15, half = lane >> 4;
    const int wm   = wav >> 1;                   // 0..3  (M sub-block of 32)
    const int wn   = wav & 1;                    // 0..1  (N sub-block of 64)

    const int m0 = (blockIdx.x >> 3) * 128;      // 64 M-blocks
    const int n0 = (blockIdx.x & 7) * 128;       // 8 N-blocks

    f32x8 acc[2][4];
#pragma unroll
    for (int r = 0; r < 2; ++r)
#pragma unroll
        for (int t = 0; t < 4; ++t)
#pragma unroll
            for (int i = 0; i < 8; ++i) acc[r][t][i] = 0.0f;

    // ---- stage k0 = 0 ----
#pragma unroll
    for (int c = 0; c < 2; ++c) {
        const int ch = tid + c * 256;            // 512 16B-chunks per tile
        const int row = ch >> 2, col = (ch & 3) * 8;
        copy16(A + (long)(m0 + row) * D_MODEL + col, &As[0][row * 32 + col]);
        copy16(W + (long)(n0 + row) * D_MODEL + col, &Bs[0][row * 32 + col]);
    }
    async_fence();
    __syncthreads();

    int p = 0;
    for (int k0 = 0; k0 < D_MODEL; k0 += 32) {
        const int k1 = k0 + 32;
        if (k1 < D_MODEL) {                      // stage next K-slab into buf[p^1]
#pragma unroll
            for (int c = 0; c < 2; ++c) {
                const int ch = tid + c * 256;
                const int row = ch >> 2, col = (ch & 3) * 8;
                copy16(A + (long)(m0 + row) * D_MODEL + k1 + col,
                       &As[p ^ 1][row * 32 + col]);
                copy16(W + (long)(n0 + row) * D_MODEL + k1 + col,
                       &Bs[p ^ 1][row * 32 + col]);
            }
        }

        // ---- compute from buf[p] ----
        BF16Frag af[2], bf[4];
#pragma unroll
        for (int r = 0; r < 2; ++r) {
            const unsigned short* ap = &As[p][(wm * 32 + r * 16 + l16) * 32];
            af[r].q[0] = ld16B(ap + half * 8);
            af[r].q[1] = ld16B(ap + half * 8 + 16);
        }
#pragma unroll
        for (int t = 0; t < 4; ++t) {
            const unsigned short* bp = &Bs[p][(wn * 64 + t * 16 + l16) * 32];
            bf[t].q[0] = ld16B(bp + half * 8);
            bf[t].q[1] = ld16B(bp + half * 8 + 16);
        }
#pragma unroll
        for (int r = 0; r < 2; ++r)
#pragma unroll
            for (int t = 0; t < 4; ++t)
                acc[r][t] = __builtin_amdgcn_wmma_f32_16x16x32_bf16(
                    false, af[r].v, false, bf[t].v, (short)0, acc[r][t], false, false);

        async_fence();
        __syncthreads();
        p ^= 1;
    }

    // ---- epilogue: C layout lane%16 = N col, VGPR i = row (i + 8*half) ----
#pragma unroll
    for (int r = 0; r < 2; ++r) {
#pragma unroll
        for (int t = 0; t < 4; ++t) {
#pragma unroll
            for (int i = 0; i < 8; ++i) {
                const int m = m0 + wm * 32 + r * 16 + i + 8 * half;
                const int n = n0 + wn * 64 + t * 16 + l16;
                const float v = acc[r][t][i];
                if (mode == 2) {
                    outf[(long)m * D_MODEL + n] = v;
                } else {
                    const int b = m >> 11, s = m & (SEQ - 1);
                    const int h = n >> 6,  d = n & (DK - 1);
                    long idx;
                    if (mode == 0)   // [B,H,S,dk]
                        idx = (((long)(b * NUM_HEADS + h)) * SEQ + s) * DK + d;
                    else             // [B,H,dk,S]
                        idx = (((long)(b * NUM_HEADS + h)) * DK + d) * SEQ + s;
                    outb[idx] = f2bf(v);
                }
            }
        }
    }
}

// ---------------------------------------------------------------------------
// Flash attention.  One block = 8 consecutive 16-query tiles of one (b,h);
// K/V tiles (64 kv x 64 d) staged once per block in double-buffered LDS.
// Per 64-wide KV step each wave runs 8 score-WMMAs + 8 PV-WMMAs around one
// online-softmax round.  Causal guard is wave-uniform (scalar branch), so
// EXEC is all-ones at every WMMA.  Q:[B,H,S,dk]  K:[B,H,S,dk]  Vt:[B,H,dk,S]
// Output O:[B,S,D] bf16.
// ---------------------------------------------------------------------------
__global__ void __launch_bounds__(256)
flash_attn(const unsigned short* __restrict__ Q,
           const unsigned short* __restrict__ K,
           const unsigned short* __restrict__ Vt,
           unsigned short* __restrict__ O) {
    __shared__ unsigned short Ks[2][64 * 64];    // 8 KB per stage
    __shared__ unsigned short Vs[2][64 * 64];    // 8 KB per stage
    __shared__ unsigned short Pb[8][16 * 64];    // per-wave P staging (C->A relayout)

    const int tid  = threadIdx.x;
    const int lane = tid & 31;
    const int wav  = __builtin_amdgcn_readfirstlane(tid >> 5);
    const int l16  = lane & 15, half = lane >> 4;

    const int bh  = blockIdx.x >> 4;             // 64 (b,h) pairs
    const int q0b = (blockIdx.x & 15) * 128;     // block covers 128 queries
    const int b   = bh >> 4, h = bh & 15;
    const int q0  = q0b + wav * 16;              // this wave's query tile

    const unsigned short* Kb = K  + (long)bh * SEQ * DK;
    const unsigned short* Vb = Vt + (long)bh * DK * SEQ;

    // Q fragment: rows q0..q0+15, d_k = 64 -> two 16x32 A fragments in regs
    const unsigned short* Qrow = Q + ((long)bh * SEQ + q0 + l16) * DK;
    BF16Frag qa0, qa1;
    qa0.q[0] = ld16B(Qrow + half * 8);
    qa0.q[1] = ld16B(Qrow + half * 8 + 16);
    qa1.q[0] = ld16B(Qrow + 32 + half * 8);
    qa1.q[1] = ld16B(Qrow + 32 + half * 8 + 16);

    f32x8 o[4];
    float mrow[8], lrow[8];
#pragma unroll
    for (int t = 0; t < 4; ++t)
#pragma unroll
        for (int i = 0; i < 8; ++i) o[t][i] = 0.0f;
#pragma unroll
    for (int i = 0; i < 8; ++i) { mrow[i] = -1e30f; lrow[i] = 0.0f; }

    unsigned short* P = &Pb[wav][0];
    const int my_jend  = q0 + 15;                // wave causal bound
    const int blk_jend = q0b + 127;              // block-uniform trip bound

    // ---- stage KV slab j0 = 0 ----
#pragma unroll
    for (int c = 0; c < 2; ++c) {
        const int ch = tid + c * 256;            // 512 16B-chunks per tile
        const int row = ch >> 3, col = (ch & 7) * 8;
        copy16(Kb + (long)row * DK + col,        &Ks[0][row * 64 + col]);
        copy16(Vb + (long)row * SEQ + col,       &Vs[0][row * 64 + col]);
    }
    async_fence();
    __syncthreads();

    int p = 0;
    for (int j0 = 0; j0 <= blk_jend; j0 += 64) {
        const int jn = j0 + 64;
        if (jn <= blk_jend) {                    // stage next slab into buf[p^1]
#pragma unroll
            for (int c = 0; c < 2; ++c) {
                const int ch = tid + c * 256;
                const int row = ch >> 3, col = (ch & 7) * 8;
                copy16(Kb + (long)(jn + row) * DK + col, &Ks[p ^ 1][row * 64 + col]);
                copy16(Vb + (long)row * SEQ + jn + col,  &Vs[p ^ 1][row * 64 + col]);
            }
        }

        if (j0 <= my_jend) {                     // wave-uniform causal guard
            // ---- scores: four 16x16 column sub-tiles, K-dim = 64 ----
            f32x8 c4[4];
#pragma unroll
            for (int t = 0; t < 4; ++t) {
                const unsigned short* Kr = &Ks[p][(t * 16 + l16) * 64];
                BF16Frag b0, b1;
                b0.q[0] = ld16B(Kr + half * 8);
                b0.q[1] = ld16B(Kr + half * 8 + 16);
                b1.q[0] = ld16B(Kr + 32 + half * 8);
                b1.q[1] = ld16B(Kr + 32 + half * 8 + 16);
                f32x8 z;
#pragma unroll
                for (int i = 0; i < 8; ++i) z[i] = 0.0f;
                c4[t] = __builtin_amdgcn_wmma_f32_16x16x32_bf16(
                    false, qa0.v, false, b0.v, (short)0, z, false, false);
                c4[t] = __builtin_amdgcn_wmma_f32_16x16x32_bf16(
                    false, qa1.v, false, b1.v, (short)0, c4[t], false, false);
            }

            // ---- scale + causal mask ----
#pragma unroll
            for (int t = 0; t < 4; ++t) {
                const int j = j0 + t * 16 + l16;
#pragma unroll
                for (int i = 0; i < 8; ++i) {
                    const int q = q0 + i + 8 * half;
                    c4[t][i] = (j > q) ? -1e30f : c4[t][i] * 0.125f;  // 1/sqrt(64)
                }
            }

            // ---- online softmax (row's 16 cols live in one 16-lane half) ----
#pragma unroll
            for (int i = 0; i < 8; ++i) {
                float rm = fmaxf(fmaxf(c4[0][i], c4[1][i]), fmaxf(c4[2][i], c4[3][i]));
                rm = fmaxf(rm, __shfl_xor(rm, 1, 32));
                rm = fmaxf(rm, __shfl_xor(rm, 2, 32));
                rm = fmaxf(rm, __shfl_xor(rm, 4, 32));
                rm = fmaxf(rm, __shfl_xor(rm, 8, 32));
                const float mn = fmaxf(mrow[i], rm);
                const float al = __expf(mrow[i] - mn);
                mrow[i] = mn;
                float rs = 0.0f;
#pragma unroll
                for (int t = 0; t < 4; ++t) {
                    c4[t][i] = __expf(c4[t][i] - mn);
                    rs += c4[t][i];
                }
                rs += __shfl_xor(rs, 1, 32);
                rs += __shfl_xor(rs, 2, 32);
                rs += __shfl_xor(rs, 4, 32);
                rs += __shfl_xor(rs, 8, 32);
                lrow[i] = lrow[i] * al + rs;
#pragma unroll
                for (int dt = 0; dt < 4; ++dt) o[dt][i] *= al;
            }

            // ---- P: C layout -> LDS -> two A fragments (intra-wave fence) ----
#pragma unroll
            for (int t = 0; t < 4; ++t)
#pragma unroll
                for (int i = 0; i < 8; ++i)
                    P[(i + 8 * half) * 64 + t * 16 + l16] = f2bf(c4[t][i]);
            asm volatile("s_wait_dscnt 0x0" ::: "memory");
            BF16Frag pa0, pa1;
            pa0.q[0] = ld16B(P + l16 * 64 + half * 8);
            pa0.q[1] = ld16B(P + l16 * 64 + half * 8 + 16);
            pa1.q[0] = ld16B(P + l16 * 64 + 32 + half * 8);
            pa1.q[1] = ld16B(P + l16 * 64 + 32 + half * 8 + 16);

            // ---- O += P @ V ----
#pragma unroll
            for (int dt = 0; dt < 4; ++dt) {
                const unsigned short* Vr = &Vs[p][(dt * 16 + l16) * 64];
                BF16Frag bv0, bv1;
                bv0.q[0] = ld16B(Vr + half * 8);
                bv0.q[1] = ld16B(Vr + half * 8 + 16);
                bv1.q[0] = ld16B(Vr + 32 + half * 8);
                bv1.q[1] = ld16B(Vr + 32 + half * 8 + 16);
                o[dt] = __builtin_amdgcn_wmma_f32_16x16x32_bf16(
                    false, pa0.v, false, bv0.v, (short)0, o[dt], false, false);
                o[dt] = __builtin_amdgcn_wmma_f32_16x16x32_bf16(
                    false, pa1.v, false, bv1.v, (short)0, o[dt], false, false);
            }
        }

        async_fence();
        __syncthreads();
        p ^= 1;
    }

    // ---- normalize and store as [B,S,D] bf16 ----
#pragma unroll
    for (int i = 0; i < 8; ++i) {
        const float inv = 1.0f / lrow[i];
        const int qg = q0 + i + 8 * half;
        const long rowoff = ((long)b * SEQ + qg) * D_MODEL + h * DK;
#pragma unroll
        for (int dt = 0; dt < 4; ++dt)
            O[rowoff + dt * 16 + l16] = f2bf(o[dt][i] * inv);
    }
}

// ---------------------------------------------------------------------------
// Host-side orchestration
// ---------------------------------------------------------------------------
extern "C" void kernel_launch(void* const* d_in, const int* in_sizes, int n_in,
                              void* d_out, int out_size, void* d_ws, size_t ws_size,
                              hipStream_t stream) {
    (void)in_sizes; (void)n_in; (void)out_size; (void)ws_size;
    const float* x  = (const float*)d_in[0];
    const float* Wq = (const float*)d_in[1];
    const float* Wk = (const float*)d_in[2];
    const float* Wv = (const float*)d_in[3];
    const float* Wo = (const float*)d_in[4];
    float* out = (float*)d_out;

    // Workspace carve-up (bf16 buffers, ~88 MB total)
    char* ws = (char*)d_ws;
    unsigned short* Xbf = (unsigned short*)ws; ws += (size_t)M_TOTAL * D_MODEL * 2;
    unsigned short* Wqb = (unsigned short*)ws; ws += (size_t)D_MODEL * D_MODEL * 2;
    unsigned short* Wkb = (unsigned short*)ws; ws += (size_t)D_MODEL * D_MODEL * 2;
    unsigned short* Wvb = (unsigned short*)ws; ws += (size_t)D_MODEL * D_MODEL * 2;
    unsigned short* Wob = (unsigned short*)ws; ws += (size_t)D_MODEL * D_MODEL * 2;
    unsigned short* Qbf = (unsigned short*)ws; ws += (size_t)M_TOTAL * D_MODEL * 2;
    unsigned short* Kbf = (unsigned short*)ws; ws += (size_t)M_TOTAL * D_MODEL * 2;
    unsigned short* Vtb = (unsigned short*)ws; ws += (size_t)M_TOTAL * D_MODEL * 2;
    unsigned short* Obf = (unsigned short*)ws; ws += (size_t)M_TOTAL * D_MODEL * 2;

    const int nX = M_TOTAL * D_MODEL;     // 8388608
    const int nW = D_MODEL * D_MODEL;     // 1048576
    cvt_f32_bf16<<<nX / 8 / 256, 256, 0, stream>>>(x,  Xbf, nX);
    cvt_f32_bf16<<<nW / 8 / 256, 256, 0, stream>>>(Wq, Wqb, nW);
    cvt_f32_bf16<<<nW / 8 / 256, 256, 0, stream>>>(Wk, Wkb, nW);
    cvt_f32_bf16<<<nW / 8 / 256, 256, 0, stream>>>(Wv, Wvb, nW);
    cvt_f32_bf16<<<nW / 8 / 256, 256, 0, stream>>>(Wo, Wob, nW);

    dim3 grid(512), blk(256);             // 64 x 8 block tiles of 128x128
    gemm_xwt<<<grid, blk, 0, stream>>>(Xbf, Wqb, Qbf, nullptr, 0);   // Q [B,H,S,dk]
    gemm_xwt<<<grid, blk, 0, stream>>>(Xbf, Wkb, Kbf, nullptr, 0);   // K [B,H,S,dk]
    gemm_xwt<<<grid, blk, 0, stream>>>(Xbf, Wvb, Vtb, nullptr, 1);   // V [B,H,dk,S]

    flash_attn<<<dim3(1024), blk, 0, stream>>>(Qbf, Kbf, Vtb, Obf);  // O [B,S,D]

    gemm_xwt<<<grid, blk, 0, stream>>>(Obf, Wob, nullptr, out, 2);   // out = O @ Wo^T
}
